// ScaleConvLayer_60765197304311
// MI455X (gfx1250) — matrix operand order repaired and checked by reading, not compiled, not run on hardware
//
#include <hip/hip_runtime.h>
#include <hip/hip_bf16.h>
#include <math.h>

// ---------------------------------------------------------------------------
// Problem constants (B=4, C=64, H=W=256)
// ---------------------------------------------------------------------------
#define HW     65536          // 256*256
#define CTOT   64
#define NTOT   16777216L      // B*64*HW
#define N4TOT  1048576L       // B*4*HW

typedef __attribute__((ext_vector_type(16))) __bf16 v16bf;
typedef __attribute__((ext_vector_type(8)))  float  v8f;

union BFrag { v16bf v; unsigned short u[16]; };

// float -> bf16 (round-to-nearest-even), as raw u16 so no scalar bf16 math is needed
static __device__ __forceinline__ unsigned short f2bf(float f) {
    unsigned u = __float_as_uint(f);
    return (unsigned short)((u + 0x7FFFu + ((u >> 16) & 1u)) >> 16);
}

// ---------------------------------------------------------------------------
// 1x1 conv 64->32 as WMMA GEMM.  Block: M=64 (spatial), N=32 (out ch).
// 8 waves -> 4 M-tiles x 2 N-tiles of 16x16.  K=64 (two x32 WMMA steps).
// out channel range [out_ch_base, out_ch_base+32) inside a 64-channel tensor.
// ---------------------------------------------------------------------------
__global__ __launch_bounds__(256) void pw_conv64_32(
    const float* __restrict__ in,    // [B][64][HW]
    const float* __restrict__ w,     // [32][64]
    const float* __restrict__ bias,  // [32]
    float* __restrict__ out,         // [B][64][HW]
    int out_ch_base)
{
    __shared__ unsigned short Alds[64][66];   // [m][k], +2 pad -> conflict-free transposed fill
    __shared__ unsigned short Blds[32][66];   // [n][k]

    int  t  = threadIdx.x;
    long m0 = (long)blockIdx.x * 64;          // global M = B*HW
    int  b  = (int)(m0 >> 16);
    int  p0 = (int)(m0 & (HW - 1));
    const float* inb = in + (long)b * CTOT * HW;

    // Stage A: 64m x 64k (16 elems/thread); contiguous per-k rows of 64 floats.
#pragma unroll
    for (int i = 0; i < 16; ++i) {
        int idx = t + i * 256;
        int k = idx >> 6, r = idx & 63;
        Alds[r][k] = f2bf(inb[(long)k * HW + p0 + r]);
    }
    // Stage B: 32n x 64k
#pragma unroll
    for (int i = 0; i < 8; ++i) {
        int idx = t + i * 256;
        int n = idx >> 6, k = idx & 63;
        Blds[n][k] = f2bf(w[n * 64 + k]);
    }
    __syncthreads();

    int wave = t >> 5, lane = t & 31;
    int mt   = (wave & 3) * 16;      // M tile base
    int nt   = (wave >> 2) * 16;     // N tile base (0 or 16)
    int rc   = lane & 15;            // A-row / B-col / D-col
    int half = lane >> 4;

    v8f acc = {};
#pragma unroll
    for (int kk = 0; kk < 2; ++kk) {
        BFrag fa, fb;
        int ka = kk * 32 + half * 16;
#pragma unroll
        for (int i = 0; i < 8; ++i) {
            fa.u[i]     = Alds[mt + rc][ka + i];
            fa.u[8 + i] = Alds[mt + rc][ka + 8 + i];
        }
        int kb = kk * 32 + half * 16;
#pragma unroll
        for (int i = 0; i < 16; ++i) fb.u[i] = Blds[nt + rc][kb + i];
        acc = __builtin_amdgcn_wmma_f32_16x16x32_bf16(
            false, fa.v, false, fb.v, (short)0, acc, false, false);
    }

    int   ch = out_ch_base + nt + rc;
    float bv = bias[nt + rc];
    float* op = out + ((long)b * CTOT + ch) * HW + p0 + mt + half * 8;
#pragma unroll
    for (int j = 0; j < 8; ++j) op[j] = acc[j] + bv;
}

// ---------------------------------------------------------------------------
// 3x3 "same" conv, Cin->64, implicit GEMM with WMMA.
// Block: M=32 (one row segment, x0..x0+31), N=64 (all out ch).
// 8 waves = 2 M-tiles x 4 N-tiles.  K ordered (tap-major, channel-minor),
// chunked by 32 with zero fill past Ktot.
// ---------------------------------------------------------------------------
__global__ __launch_bounds__(256) void conv3x3_wmma(
    const float* __restrict__ in,    // [B][Cin][256][256]
    const float* __restrict__ w,     // [64][Cin][3][3]
    const float* __restrict__ bias,  // [64]
    float* __restrict__ out,         // [B][64][256][256]
    int Cin, int CinShift, int Kchunks)
{
    __shared__ unsigned short Ach[32][34];
    __shared__ unsigned short Bch[64][34];

    int t    = threadIdx.x;
    int blk  = blockIdx.x;                 // B * 256 * 8
    int x0   = (blk & 7) * 32;
    int y    = (blk >> 3) & 255;
    int b    = blk >> 11;
    int Ktot = Cin * 9;
    int cm   = Cin - 1;

    int wave = t >> 5, lane = t & 31;
    int mt   = (wave & 1) * 16;
    int nt   = (wave >> 1) * 16;
    int rc   = lane & 15;
    int half = lane >> 4;

    v8f acc = {};
    for (int kk = 0; kk < Kchunks; ++kk) {
        int kbase = kk * 32;
        // Stage A chunk: 32m x 32k im2col slice (4 elems/thread)
#pragma unroll
        for (int i = 0; i < 4; ++i) {
            int idx  = t + i * 256;
            int kloc = idx >> 5, m = idx & 31;
            int kg   = kbase + kloc;
            float v  = 0.0f;
            if (kg < Ktot) {
                int tap = kg >> CinShift, c = kg & cm;
                int dy = tap / 3 - 1, dx = tap - (tap / 3) * 3 - 1;
                int yy = y + dy, xx = x0 + m + dx;
                if (yy >= 0 && yy < 256 && xx >= 0 && xx < 256)
                    v = in[(((long)b * Cin + c) * 256 + yy) * 256 + xx];
            }
            Ach[m][kloc] = f2bf(v);
        }
        // Stage B chunk: 64n x 32k (8 elems/thread)
#pragma unroll
        for (int i = 0; i < 8; ++i) {
            int idx  = t + i * 256;
            int n = idx >> 5, kloc = idx & 31;
            int kg = kbase + kloc;
            float v = 0.0f;
            if (kg < Ktot) {
                int tap = kg >> CinShift, c = kg & cm;
                v = w[(n * Cin + c) * 9 + tap];
            }
            Bch[n][kloc] = f2bf(v);
        }
        // Prefetch next weight chunk (gfx1250 global_prefetch)
        int kng = kbase + 32;
        if (kng < Ktot)
            __builtin_prefetch(&w[((t >> 2) * Cin + (kng & cm)) * 9 + (kng >> CinShift)], 0, 1);
        __syncthreads();

        BFrag fa, fb;
        int ka = half * 16;
#pragma unroll
        for (int i = 0; i < 8; ++i) {
            fa.u[i]     = Ach[mt + rc][ka + i];
            fa.u[8 + i] = Ach[mt + rc][ka + 8 + i];
        }
        int kb = half * 16;
#pragma unroll
        for (int i = 0; i < 16; ++i) fb.u[i] = Bch[nt + rc][kb + i];
        acc = __builtin_amdgcn_wmma_f32_16x16x32_bf16(
            false, fa.v, false, fb.v, (short)0, acc, false, false);
        __syncthreads();
    }

    int   ch = nt + rc;
    float bv = bias[ch];
    float* op = out + (((long)b * 64 + ch) * 256 + y) * 256 + x0 + mt + half * 8;
#pragma unroll
    for (int j = 0; j < 8; ++j) op[j] = acc[j] + bv;
}

// ---------------------------------------------------------------------------
// Depthwise 3x3 (32 channels) inside 64-channel tensors
// ---------------------------------------------------------------------------
__global__ __launch_bounds__(256) void dwconv3x3(
    const float* __restrict__ in, const float* __restrict__ w,
    const float* __restrict__ bias, float* __restrict__ out,
    int in_ch_base, int out_ch_base)
{
    long i = (long)blockIdx.x * 256 + threadIdx.x;   // B*32*HW
    int p = (int)(i & (HW - 1));
    int c = (int)((i >> 16) & 31);
    int b = (int)(i >> 21);
    int y = p >> 8, x = p & 255;
    const float* src = in + ((long)(b * CTOT + in_ch_base + c)) * HW;
    float s = bias[c];
#pragma unroll
    for (int dy = -1; dy <= 1; ++dy)
#pragma unroll
        for (int dx = -1; dx <= 1; ++dx) {
            int yy = y + dy, xx = x + dx;
            if (yy >= 0 && yy < 256 && xx >= 0 && xx < 256)
                s += w[c * 9 + (dy + 1) * 3 + (dx + 1)] * src[yy * 256 + xx];
        }
    out[((long)(b * CTOT + out_ch_base + c)) * HW + p] = s;
}

// ---------------------------------------------------------------------------
// Elementwise helpers
// ---------------------------------------------------------------------------
__global__ __launch_bounds__(256) void lrelu_ip(float* x, float alpha, long n) {
    long i = (long)blockIdx.x * 256 + threadIdx.x;
    if (i < n) { float v = x[i]; x[i] = v > 0.0f ? v : alpha * v; }
}
__global__ __launch_bounds__(256) void add_res(float* out, const float* __restrict__ in, long n) {
    long i = (long)blockIdx.x * 256 + threadIdx.x;
    if (i < n) out[i] += in[i];
}

// ---------------------------------------------------------------------------
// 1x1 conv 64->4 (VALU): pm / fps (optional addend = origin)
// ---------------------------------------------------------------------------
__global__ __launch_bounds__(256) void conv1x1_c4(
    const float* __restrict__ in, const float* __restrict__ w,
    const float* __restrict__ bias, const float* __restrict__ addend,
    float* __restrict__ out)
{
    long i = (long)blockIdx.x * 256 + threadIdx.x;   // B*HW
    int b = (int)(i >> 16);
    int p = (int)(i & (HW - 1));
    const float* src = in + (long)b * CTOT * HW + p;
    float a0 = bias[0], a1 = bias[1], a2 = bias[2], a3 = bias[3];
#pragma unroll 8
    for (int k = 0; k < 64; ++k) {
        float v = src[(long)k * HW];
        a0 += v * w[k]; a1 += v * w[64 + k]; a2 += v * w[128 + k]; a3 += v * w[192 + k];
    }
    long ob = (long)b * 4 * HW + p;
    if (addend) {
        a0 += addend[ob]; a1 += addend[ob + HW];
        a2 += addend[ob + 2 * HW]; a3 += addend[ob + 3 * HW];
    }
    out[ob] = a0; out[ob + HW] = a1; out[ob + 2 * HW] = a2; out[ob + 3 * HW] = a3;
}

// ---------------------------------------------------------------------------
// Per-batch argmax over pm channel 3 -> pt (to d_out tail) + warp params
// ---------------------------------------------------------------------------
__global__ __launch_bounds__(256) void argmax_params(
    const float* __restrict__ pm, float* __restrict__ pt_out, float* __restrict__ wp)
{
    __shared__ float sv[256];
    __shared__ int   si[256];
    int b = blockIdx.x, t = threadIdx.x;
    const float* ch3 = pm + ((long)b * 4 + 3) * HW;
    float best = -INFINITY; int bi = HW;
    for (int i = t; i < HW; i += 256) {
        float v = ch3[i];
        if (v > best || (v == best && i < bi)) { best = v; bi = i; }
    }
    sv[t] = best; si[t] = bi; __syncthreads();
    for (int o = 128; o > 0; o >>= 1) {
        if (t < o) {
            if (sv[t + o] > sv[t] || (sv[t + o] == sv[t] && si[t + o] < si[t])) {
                sv[t] = sv[t + o]; si[t] = si[t + o];
            }
        }
        __syncthreads();
    }
    if (t == 0) {
        int idx = si[0];
        float p0 = pm[((long)b * 4 + 0) * HW + idx];
        float p1 = pm[((long)b * 4 + 1) * HW + idx];
        float p2 = pm[((long)b * 4 + 2) * HW + idx];
        float p3 = pm[((long)b * 4 + 3) * HW + idx];
        pt_out[b * 4 + 0] = p0; pt_out[b * 4 + 1] = p1;
        pt_out[b * 4 + 2] = p2; pt_out[b * 4 + 3] = p3;
        float ty = (1.0f / (1.0f + expf(-p0)) - 0.5f) * 256.0f;   // y-shift (Hh)
        float tx = (1.0f / (1.0f + expf(-p1)) - 0.5f) * 256.0f;   // x-shift (Ww)
        float sg = 1.0f / (1.0f + expf(-p2));
        float sc = exp2f(2.0f * (2.0f * sg - 1.0f));              // M_LOG2 = 2
        wp[b * 3 + 0] = ty; wp[b * 3 + 1] = tx; wp[b * 3 + 2] = sc;
    }
}

// ---------------------------------------------------------------------------
// Composed nearest warp: tr = warp(warp(fps, t_flow), s_flow)
// ---------------------------------------------------------------------------
__global__ __launch_bounds__(256) void warp_compose(
    const float* __restrict__ fps, const float* __restrict__ wp, float* __restrict__ out)
{
    long i = (long)blockIdx.x * 256 + threadIdx.x;   // B*HW
    int b = (int)(i >> 16);
    int p = (int)(i & (HW - 1));
    int y = p >> 8, x = p & 255;
    float ty = wp[b * 3 + 0], tx = wp[b * 3 + 1], sc = wp[b * 3 + 2];
    // outer warp (scale flow), clipped to grid
    int Y = min(255, max(0, (int)roundf((float)y + ((float)y - 127.5f) * (sc - 1.0f))));
    int X = min(255, max(0, (int)roundf((float)x + ((float)x - 127.5f) * (sc - 1.0f))));
    // inner warp (translation flow)
    int ys = min(255, max(0, (int)roundf((float)Y + ty)));
    int xs = min(255, max(0, (int)roundf((float)X + tx)));
    long src = (long)b * 4 * HW + ys * 256 + xs;
    long dst = (long)b * 4 * HW + p;
    out[dst]          = fps[src];
    out[dst + HW]     = fps[src + HW];
    out[dst + 2 * HW] = fps[src + 2 * HW];
    out[dst + 3 * HW] = fps[src + 3 * HW];
}

// ---------------------------------------------------------------------------
// Deterministic batchnorm: partial sums -> final stats -> normalize
// ---------------------------------------------------------------------------
__global__ __launch_bounds__(256) void bn_reduce(const float* __restrict__ x, float* __restrict__ part)
{
    __shared__ float ls[256], lq[256];
    int blk = blockIdx.x;                 // 64 ch * 64 slices
    int c = blk >> 6, s = blk & 63;
    int b = s >> 4, pbase = (s & 15) * 4096;
    const float* src = x + ((long)(b * CTOT + c)) * HW + pbase;
    float sum = 0.0f, sq = 0.0f;
    for (int i = threadIdx.x; i < 4096; i += 256) {
        float v = src[i]; sum += v; sq += v * v;
    }
    int t = threadIdx.x;
    ls[t] = sum; lq[t] = sq; __syncthreads();
    for (int o = 128; o > 0; o >>= 1) {
        if (t < o) { ls[t] += ls[t + o]; lq[t] += lq[t + o]; }
        __syncthreads();
    }
    if (t == 0) { part[blk * 2] = ls[0]; part[blk * 2 + 1] = lq[0]; }
}
__global__ __launch_bounds__(64) void bn_final(const float* __restrict__ part, float* __restrict__ stats)
{
    int c = threadIdx.x;
    float sum = 0.0f, sq = 0.0f;
    for (int s = 0; s < 64; ++s) {
        sum += part[(c * 64 + s) * 2];
        sq  += part[(c * 64 + s) * 2 + 1];
    }
    stats[c] = sum; stats[64 + c] = sq;
}
__global__ __launch_bounds__(256) void bn_norm(
    const float* __restrict__ x, const float* __restrict__ stats,
    const float* __restrict__ g, const float* __restrict__ beta, float* __restrict__ out)
{
    long i = (long)blockIdx.x * 256 + threadIdx.x;   // B*64*HW
    int c = (int)((i >> 16) & 63);
    float m = stats[c] * (1.0f / 262144.0f);
    float v = stats[64 + c] * (1.0f / 262144.0f) - m * m;
    out[i] = (x[i] - m) * rsqrtf(v + 1e-5f) * g[c] + beta[c];
}

// ---------------------------------------------------------------------------
// Host launcher
// ---------------------------------------------------------------------------
extern "C" void kernel_launch(void* const* d_in, const int* in_sizes, int n_in,
                              void* d_out, int out_size, void* d_ws, size_t ws_size,
                              hipStream_t stream)
{
    (void)in_sizes; (void)n_in; (void)out_size; (void)ws_size;

    auto W = [&](int i) { return (const float*)d_in[i]; };
    const float* x      = W(0);
    const float* origin = W(1);
    // params flat order: gb1(2..9) gb2(10..17) w_ex(18) b_ex(19) bn1_g(20) bn1_b(21)
    //   w_reg(22) b_reg(23) w_fpt(24) b_fpt(25) w3(26) b3(27) gb4(28..35) bn2_g(36) bn2_b(37)

    float* ws  = (float*)d_ws;
    float* T1  = ws; ws += NTOT;
    float* H1  = ws; ws += NTOT;
    float* H2  = ws; ws += NTOT;
    float* PM  = ws; ws += N4TOT;
    float* FPS = ws; ws += N4TOT;
    float* TR  = ws; ws += N4TOT;
    float* PART = ws; ws += 8192;
    float* STATS = ws; ws += 128;
    float* WP  = ws; ws += 16;

    float* out_main = (float*)d_out;
    float* pt_out   = out_main + NTOT;

    // ghost bottleneck: in -> ob   (temp t1)
    auto gb = [&](const float* inb, float* ob, float* t1, int wbase) {
        pw_conv64_32<<<4096, 256, 0, stream>>>(inb, W(wbase + 0), W(wbase + 1), t1, 0);
        dwconv3x3  <<<32768, 256, 0, stream>>>(t1, W(wbase + 2), W(wbase + 3), t1, 0, 32);
        lrelu_ip   <<<65536, 256, 0, stream>>>(t1, 0.0f, NTOT);                       // relu(concat)
        pw_conv64_32<<<4096, 256, 0, stream>>>(t1, W(wbase + 4), W(wbase + 5), ob, 0);
        dwconv3x3  <<<32768, 256, 0, stream>>>(ob, W(wbase + 6), W(wbase + 7), ob, 0, 32);
        add_res    <<<65536, 256, 0, stream>>>(ob, inb, NTOT);                        // residual
    };

    // h = leaky_relu(gb2(gb1(x)))
    gb(x,  H1, T1, 2);
    gb(H1, H2, T1, 10);
    lrelu_ip<<<65536, 256, 0, stream>>>(H2, 0.01f, NTOT);

    // xe = bn1(conv3x3(x, w_ex))   (into H1)
    conv3x3_wmma<<<8192, 256, 0, stream>>>(x, W(18), W(19), H1, 64, 6, 18);
    bn_reduce<<<4096, 256, 0, stream>>>(H1, PART);
    bn_final <<<1, 64, 0, stream>>>(PART, STATS);
    bn_norm  <<<65536, 256, 0, stream>>>(H1, STATS, W(20), W(21), H1);

    // pm = conv1x1(xe, w_reg); pt + warp params
    conv1x1_c4<<<1024, 256, 0, stream>>>(H1, W(22), W(23), nullptr, PM);
    argmax_params<<<4, 256, 0, stream>>>(PM, pt_out, WP);

    // fps = conv1x1(h, w_fpt) + origin; tr = composed double warp
    conv1x1_c4<<<1024, 256, 0, stream>>>(H2, W(24), W(25), origin, FPS);
    warp_compose<<<1024, 256, 0, stream>>>(FPS, WP, TR);

    // out3 = conv3x3(tr, w3)  (into H2, Cin=4, K=36 -> 2 chunks)
    conv3x3_wmma<<<8192, 256, 0, stream>>>(TR, W(26), W(27), H2, 4, 2, 2);

    // out = bn2(gb4(out3)) -> d_out
    gb(H2, H1, T1, 28);
    bn_reduce<<<4096, 256, 0, stream>>>(H1, PART);
    bn_final <<<1, 64, 0, stream>>>(PART, STATS);
    bn_norm  <<<65536, 256, 0, stream>>>(H1, STATS, W(36), W(37), out_main);
}
